// SwitchFeedForward_80075370267319
// MI455X (gfx1250) — matrix operand order repaired and checked by reading, not compiled, not run on hardware
//
#include <hip/hip_runtime.h>
#include <hip/hip_bf16.h>

typedef __bf16 bf16_t;
typedef __attribute__((ext_vector_type(16))) __bf16 v16bf;
typedef __attribute__((ext_vector_type(8)))  __bf16 v8bf;
typedef __attribute__((ext_vector_type(4)))  __bf16 v4bf;
typedef __attribute__((ext_vector_type(8)))  float  v8f;
typedef int gv4i __attribute__((vector_size(16)));   // matches builtin's V4i param

#define T_TOK   8192
#define DMODEL  1024
#define NEXP    8
#define CHUNK   128
#define NCHUNK  64              // T_TOK / CHUNK
#define SUPER   32              // rows per GEMM supertile
#define MAXTILES 264            // ceil((T_TOK + NEXP*31)/32)
#define MAXROWS (MAXTILES*SUPER)
#define KSLICE  256             // K columns staged per LDS buffer
#define LDA     264             // KSLICE + 8 pad -> 132-dword row stride (bank-friendly)

#define AS1 __attribute__((address_space(1)))
#define AS3 __attribute__((address_space(3)))

#if defined(__has_builtin)
#if __has_builtin(__builtin_amdgcn_global_load_async_to_lds_b128) && \
    __has_builtin(__builtin_amdgcn_s_wait_asynccnt)
#define USE_ASYNC_LDS 1
#endif
#endif
#ifndef USE_ASYNC_LDS
#define USE_ASYNC_LDS 0
#endif

union bfpack { v16bf v; v8bf p[2]; };

// ---------------- Router: logits -> softmax -> top1 ----------------
__global__ __launch_bounds__(256) void router_kernel(
    const float* __restrict__ x, const float* __restrict__ Ws,
    const float* __restrict__ bs, int* __restrict__ routes,
    float* __restrict__ pmax) {
  __shared__ float slog[NEXP];
  const int t = blockIdx.x;
  const int tid = threadIdx.x;
  const int wave = tid >> 5, lane = tid & 31;
  const float* xt = x + (size_t)t * DMODEL;
  const float* w  = Ws + (size_t)wave * DMODEL;
  float s = 0.f;
  for (int j = lane; j < DMODEL; j += 32) s += xt[j] * w[j];
  #pragma unroll
  for (int off = 16; off > 0; off >>= 1) s += __shfl_xor(s, off, 32);
  if (lane == 0) slog[wave] = s + bs[wave];
  __syncthreads();
  if (tid == 0) {
    float m = slog[0]; int arg = 0;
    #pragma unroll
    for (int e = 1; e < NEXP; ++e) if (slog[e] > m) { m = slog[e]; arg = e; } // first-max
    float sum = 0.f;
    #pragma unroll
    for (int e = 0; e < NEXP; ++e) sum += __expf(slog[e] - m);
    routes[t] = arg;
    pmax[t]   = 1.0f / sum;   // exp(lmax-m)/sum with lmax==m
  }
}

// ---------------- Per-chunk counts + stable in-chunk ranks ----------------
__global__ __launch_bounds__(CHUNK) void count_kernel(
    const int* __restrict__ routes, int* __restrict__ rank, int* __restrict__ cnt) {
  __shared__ int sr[CHUNK];
  __shared__ int sc[NEXP];
  const int b = blockIdx.x, i = threadIdx.x;
  const int t = b * CHUNK + i;
  if (i < NEXP) sc[i] = 0;
  sr[i] = routes[t];
  __syncthreads();
  const int e = sr[i];
  int r = 0;
  for (int j = 0; j < i; ++j) r += (sr[j] == e) ? 1 : 0;
  rank[t] = r;
  atomicAdd(&sc[e], 1);
  __syncthreads();
  if (i < NEXP) cnt[b * NEXP + i] = sc[i];
}

// ---------------- Scan: offsets, 32-padded segments, supertile->expert map ----------------
__global__ __launch_bounds__(256) void scan_kernel(
    const int* __restrict__ cnt, int* __restrict__ prefix,
    int* __restrict__ start, int* __restrict__ pstart,
    int* __restrict__ tileExpert, int* __restrict__ rowmap) {
  const int tid = threadIdx.x;
  for (int i = tid; i < MAXROWS; i += 256) rowmap[i] = -1;
  if (tid == 0) {
    int tot[NEXP];
    #pragma unroll
    for (int e = 0; e < NEXP; ++e) tot[e] = 0;
    for (int b = 0; b < NCHUNK; ++b)
      for (int e = 0; e < NEXP; ++e) { prefix[b*NEXP+e] = tot[e]; tot[e] += cnt[b*NEXP+e]; }
    int us = 0, ps = 0;
    for (int e = 0; e < NEXP; ++e) {
      start[e] = us; pstart[e] = ps;
      const int pt = (tot[e] + (SUPER-1)) & ~(SUPER-1);
      for (int tl = ps / SUPER; tl < (ps + pt) / SUPER; ++tl) tileExpert[tl] = e;
      us += tot[e]; ps += pt;
    }
    for (int tl = ps / SUPER; tl < MAXTILES; ++tl) tileExpert[tl] = -1;
  }
}

// ---------------- Scatter tokens to padded sorted rows, gather x*pmax -> bf16 ----------------
__global__ __launch_bounds__(256) void scatter_kernel(
    const float* __restrict__ x, const int* __restrict__ routes,
    const int* __restrict__ rank, const float* __restrict__ pmax,
    const int* __restrict__ prefix, const int* __restrict__ start,
    const int* __restrict__ pstart, int* __restrict__ rowmap,
    bf16_t* __restrict__ xs) {
  __shared__ int   spos[CHUNK];
  __shared__ float sp[CHUNK];
  const int b = blockIdx.x, tid = threadIdx.x;
  const int t0 = b * CHUNK;
  if (tid < CHUNK) {
    const int t = t0 + tid;
    const int e = routes[t];
    const int r = prefix[b*NEXP+e] + rank[t];
    const int p = pstart[e] + r;
    rowmap[p] = start[e] + r;    // padded row -> sorted output row
    spos[tid] = p;
    sp[tid]   = pmax[t];
  }
  __syncthreads();
  const int d0 = tid * 4;
  for (int lt = 0; lt < CHUNK; ++lt) {
    const float4 v = *(const float4*)(x + (size_t)(t0 + lt) * DMODEL + d0);
    const float sc = sp[lt];
    v4bf o = { (bf16_t)(v.x*sc), (bf16_t)(v.y*sc), (bf16_t)(v.z*sc), (bf16_t)(v.w*sc) };
    *(v4bf*)(xs + (size_t)spos[lt] * DMODEL + d0) = o;
  }
}

// ---------------- Transpose-convert fp32 [k][n] -> bf16 [n][k] ----------------
__global__ __launch_bounds__(256) void convert_kernel(
    const float* __restrict__ W1, const float* __restrict__ W2,
    bf16_t* __restrict__ W1t, bf16_t* __restrict__ W2t) {
  __shared__ float tile[32][33];
  const int kt = blockIdx.x & 31, nt = blockIdx.x >> 5;
  const int e = blockIdx.y;
  const float* src = (blockIdx.z ? W2 : W1) + (size_t)e * DMODEL * DMODEL;
  bf16_t*      dst = (blockIdx.z ? W2t : W1t) + (size_t)e * DMODEL * DMODEL;
  const int tx = threadIdx.x & 31, ty = threadIdx.x >> 5;
  #pragma unroll
  for (int r = 0; r < 4; ++r)
    tile[ty + 8*r][tx] = src[(size_t)(kt*32 + ty + 8*r) * DMODEL + (nt*32 + tx)];
  __syncthreads();
  #pragma unroll
  for (int r = 0; r < 4; ++r)
    dst[(size_t)(nt*32 + ty + 8*r) * DMODEL + (kt*32 + tx)] = (bf16_t)tile[tx][ty + 8*r];
}

// ---------------- FFN GEMM: 32x(512) block tile, 32x64 wave tile ----------------
// A (32 rows x full K) staged through double-buffered async LDS K-slices.
// IS_FC1: out = relu(A@W + b) -> bf16 ;  else: out = A@W + b -> fp32 via rowmap.
template <bool IS_FC1>
__global__ __launch_bounds__(256) void ffn_gemm_kernel(
    const bf16_t* __restrict__ A, const bf16_t* __restrict__ Wt,
    const float* __restrict__ bias, const int* __restrict__ tileExpert,
    const int* __restrict__ rowmap, bf16_t* __restrict__ hout,
    float* __restrict__ yout) {
  __shared__ bf16_t sA[2][SUPER][LDA];    // 2 x 32 x 264 bf16 = 33 KB
  const int mtile = blockIdx.x;
  const int e = tileExpert[mtile];
  if (e < 0) return;
  const int tid = threadIdx.x;
  const int wave = tid >> 5, lane = tid & 31;
  const int lane16 = lane & 15, half = lane >> 4;
  const int nbase = blockIdx.y * 512 + wave * 64;
  const int rowbase = mtile * SUPER;

  const bf16_t* Abase = A + (size_t)rowbase * DMODEL;
  const bf16_t* Bbase = Wt + (size_t)e * DMODEL * DMODEL;
  const bf16_t* Bp[4];
  #pragma unroll
  for (int i = 0; i < 4; ++i) Bp[i] = Bbase + (size_t)(nbase + 16*i + lane16) * DMODEL;

  // --- stage one 32 x KSLICE A slice into LDS buffer `buf` (async if available) ---
  auto stage = [&](int ck, int buf) {
    const bf16_t* g0 = Abase + ck * KSLICE;
    #pragma unroll
    for (int it = 0; it < 4; ++it) {
      const int idx = tid + 256 * it;          // 0..1023
      const int row = idx >> 5;                // 32 x b128 per row
      const int c8  = (idx & 31) * 8;
      const bf16_t* src = g0 + (size_t)row * DMODEL + c8;
      bf16_t* dst = &sA[buf][row][c8];
#if USE_ASYNC_LDS
      __builtin_amdgcn_global_load_async_to_lds_b128(
          (AS1 gv4i*)src, (AS3 gv4i*)dst, 0, 0);
#else
      *(v8bf*)dst = *(const v8bf*)src;
#endif
    }
  };

  v8f c[4][2] = { { {}, {} }, { {}, {} }, { {}, {} }, { {}, {} } };

  stage(0, 0);
#if USE_ASYNC_LDS
  __builtin_amdgcn_s_wait_asynccnt(0);
#endif
  __syncthreads();

  for (int ck = 0; ck < DMODEL / KSLICE; ++ck) {
    const int cur = ck & 1;
    if (ck + 1 < DMODEL / KSLICE) stage(ck + 1, cur ^ 1);   // overlap next slice
    const int kg0 = ck * KSLICE;
    for (int kl = 0; kl < KSLICE; kl += 32) {
      const int kb = kg0 + kl;
      __builtin_prefetch(Bp[0] + kb + 256, 0, 1);
      __builtin_prefetch(Bp[2] + kb + 256, 0, 1);
      bfpack a0, a1;
      a0.p[0] = *(const v8bf*)&sA[cur][lane16     ][kl + half * 8];
      a0.p[1] = *(const v8bf*)&sA[cur][lane16     ][kl + 16 + half * 8];
      a1.p[0] = *(const v8bf*)&sA[cur][16 + lane16][kl + half * 8];
      a1.p[1] = *(const v8bf*)&sA[cur][16 + lane16][kl + 16 + half * 8];
      #pragma unroll
      for (int i = 0; i < 4; ++i) {
        bfpack b;
        b.p[0] = *(const v8bf*)(Bp[i] + kb + half * 8);
        b.p[1] = *(const v8bf*)(Bp[i] + kb + 16 + half * 8);
        c[i][0] = __builtin_amdgcn_wmma_f32_16x16x32_bf16(
            false, a0.v, false, b.v, (short)0, c[i][0], false, false);
        c[i][1] = __builtin_amdgcn_wmma_f32_16x16x32_bf16(
            false, a1.v, false, b.v, (short)0, c[i][1], false, false);
      }
    }
#if USE_ASYNC_LDS
    __builtin_amdgcn_s_wait_asynccnt(0);
#endif
    __syncthreads();
  }

  float bvals[4];
  #pragma unroll
  for (int i = 0; i < 4; ++i) bvals[i] = bias[e * DMODEL + nbase + 16*i + lane16];

  if constexpr (IS_FC1) {
    #pragma unroll
    for (int i = 0; i < 4; ++i) {
      const int n = nbase + 16*i + lane16;
      #pragma unroll
      for (int r = 0; r < 2; ++r) {
        #pragma unroll
        for (int v = 0; v < 8; ++v) {
          const int row = rowbase + r * 16 + half * 8 + v;
          hout[(size_t)row * DMODEL + n] = (bf16_t)fmaxf(c[i][r][v] + bvals[i], 0.0f);
        }
      }
    }
  } else {
    #pragma unroll
    for (int r = 0; r < 2; ++r) {
      #pragma unroll
      for (int v = 0; v < 8; ++v) {
        const int row = rowbase + r * 16 + half * 8 + v;
        const int orow = rowmap[row];
        if (orow < 0) continue;           // padding row -> dropped
        #pragma unroll
        for (int i = 0; i < 4; ++i) {
          const int n = nbase + 16*i + lane16;
          yout[(size_t)orow * DMODEL + n] = c[i][r][v] + bvals[i];
        }
      }
    }
  }
}

extern "C" void kernel_launch(void* const* d_in, const int* in_sizes, int n_in,
                              void* d_out, int out_size, void* d_ws, size_t ws_size,
                              hipStream_t stream) {
  const float* x  = (const float*)d_in[0];
  const float* Ws = (const float*)d_in[1];
  const float* bs = (const float*)d_in[2];
  const float* W1 = (const float*)d_in[3];
  const float* b1 = (const float*)d_in[4];
  const float* W2 = (const float*)d_in[5];
  const float* b2 = (const float*)d_in[6];
  float* out = (float*)d_out;

  char* ws = (char*)d_ws;
  size_t off = 0;
  auto alloc = [&](size_t bytes) -> void* {
    void* p = ws + off;
    off = (off + bytes + 255) & ~(size_t)255;
    return p;
  };
  int*    routes    = (int*)   alloc((size_t)T_TOK * 4);
  int*    rank      = (int*)   alloc((size_t)T_TOK * 4);
  float*  pmax      = (float*) alloc((size_t)T_TOK * 4);
  int*    cnt       = (int*)   alloc((size_t)NCHUNK * NEXP * 4);
  int*    prefix    = (int*)   alloc((size_t)NCHUNK * NEXP * 4);
  int*    start     = (int*)   alloc((size_t)NEXP * 4);
  int*    pstart    = (int*)   alloc((size_t)NEXP * 4);
  int*    tileExp   = (int*)   alloc((size_t)MAXTILES * 4);
  int*    rowmap    = (int*)   alloc((size_t)MAXROWS * 4);
  bf16_t* xs        = (bf16_t*)alloc((size_t)MAXROWS * DMODEL * 2);
  bf16_t* hbuf      = (bf16_t*)alloc((size_t)MAXROWS * DMODEL * 2);
  bf16_t* W1t       = (bf16_t*)alloc((size_t)NEXP * DMODEL * DMODEL * 2);
  bf16_t* W2t       = (bf16_t*)alloc((size_t)NEXP * DMODEL * DMODEL * 2);
  (void)ws_size; (void)in_sizes; (void)n_in; (void)out_size;

  router_kernel <<<T_TOK, 256, 0, stream>>>(x, Ws, bs, routes, pmax);
  count_kernel  <<<NCHUNK, CHUNK, 0, stream>>>(routes, rank, cnt);
  scan_kernel   <<<1, 256, 0, stream>>>(cnt, prefix, start, pstart, tileExp, rowmap);
  scatter_kernel<<<NCHUNK, 256, 0, stream>>>(x, routes, rank, pmax, prefix, start,
                                             pstart, rowmap, xs);
  convert_kernel<<<dim3(1024, NEXP, 2), 256, 0, stream>>>(W1, W2, W1t, W2t);
  ffn_gemm_kernel<true>  <<<dim3(MAXTILES, 2), 256, 0, stream>>>(
      xs, W1t, b1, tileExp, rowmap, hbuf, nullptr);
  ffn_gemm_kernel<false> <<<dim3(MAXTILES, 2), 256, 0, stream>>>(
      hbuf, W2t, b2, tileExp, rowmap, nullptr, out);
}
